// MosAttention_83648783057406
// MI455X (gfx1250) — matrix operand order repaired and checked by reading, not compiled, not run on hardware
//
#include <hip/hip_runtime.h>
#include <hip/hip_bf16.h>

// ---------------------------------------------------------------------------
// MosAttention pipeline for MI455X (gfx1250, wave32, WMMA bf16).
//
// Algebraic restructuring (see analysis):
//   q_{e,p} = (q_w @ scatter_w[p]) @ events[e]        (same for v, g)
//   out[e]  = events[e] + sum_p (gather_w[p] @ o_w) @ (q*h)_{e,p}
// so the entire pipeline is dense GEMMs on events[16384,256] plus a cheap
// ragged gated scan in (patch,time) order, then LayerNorm.
// ---------------------------------------------------------------------------

typedef __attribute__((ext_vector_type(16))) __bf16 v16bf;
typedef __attribute__((ext_vector_type(8)))  __bf16 v8bf;
typedef __attribute__((ext_vector_type(8)))  float  v8f;

#define NE   16384              // events
#define PP   9                  // kernel positions
#define DD   256                // channels
#define MR   (NE * PP)          // patch-copy rows = 147456

// ---------------------------------------------------------------------------
// fp32 -> bf16 elementwise convert
// ---------------------------------------------------------------------------
__global__ __launch_bounds__(256) void cvt_bf16_kernel(const float* __restrict__ in,
                                                       __bf16* __restrict__ out, int n) {
    int i = blockIdx.x * 256 + threadIdx.x;
    if (i < n) out[i] = (__bf16)in[i];
}

// ---------------------------------------------------------------------------
// Weight fusion: CW[t][p][n][k] (bf16, row-major [n,k] == WMMA B layout)
//   t=0: q_w @ scatter_w[p]   t=1: v_w @ scatter_w[p]
//   t=2: g_w @ scatter_w[p]   t=3: gather_w[p] @ o_w
// grid = (256, 36): blockIdx.x = n, threadIdx.x = k  (coalesced over k)
// ---------------------------------------------------------------------------
__global__ __launch_bounds__(256) void fuse_weights_kernel(
    const float* __restrict__ qw, const float* __restrict__ vw,
    const float* __restrict__ gw, const float* __restrict__ ow,
    const float* __restrict__ scatter, const float* __restrict__ gather,
    __bf16* __restrict__ CW) {
    int tp = blockIdx.y;                 // 0..35
    int t = tp / PP, p = tp % PP;
    int n = blockIdx.x;
    int k = threadIdx.x;
    const float* Af = (t == 0) ? qw : (t == 1) ? vw : (t == 2) ? gw
                                    : (gather + (size_t)p * DD * DD);
    const float* Bf = (t < 3) ? (scatter + (size_t)p * DD * DD) : ow;
    const float* arow = Af + (size_t)n * DD;
    float acc = 0.f;
#pragma unroll 4
    for (int j = 0; j < DD; ++j)
        acc += arow[j] * Bf[(size_t)j * DD + k];
    CW[(size_t)tp * DD * DD + (size_t)n * DD + k] = (__bf16)acc;
}

// ---------------------------------------------------------------------------
// WMMA fragment helpers (v_wmma_f32_16x16x32_bf16)
//   A 16x32: lane L -> row L&15; 8 bf16 at k0+(L>>4)*8 and 8 bf16 at +16
//   B 32x16: lane L -> col L&15; 8 bf16 at k0+(L>>4)*8 and 8 bf16 at +16
//   C 16x16: lane L -> col L&15; vgpr r -> row r+(L>>4)*8
// ---------------------------------------------------------------------------
__device__ __forceinline__ v16bf load_a_frag(const __bf16* __restrict__ arow, int k0) {
    v8bf lo = *(const v8bf*)(arow + k0);
    v8bf hi = *(const v8bf*)(arow + k0 + 16);
    return __builtin_shufflevector(lo, hi, 0, 1, 2, 3, 4, 5, 6, 7,
                                           8, 9, 10, 11, 12, 13, 14, 15);
}

// ---------------------------------------------------------------------------
// Stage 1: qvg[tp][e][c] = events @ CW[tp]^T  (27 GEMMs, bf16 out, f32 acc)
// One wave per 16x64 output tile; wave id encodes {tp, mtile, ntile}.
// ---------------------------------------------------------------------------
__global__ __launch_bounds__(256) void gemm_qvg_kernel(
    const __bf16* __restrict__ evbf, const __bf16* __restrict__ CW,
    __bf16* __restrict__ qvg) {
    const int wave = (blockIdx.x << 3) | (threadIdx.x >> 5);
    const int lane = threadIdx.x & 31;
    const int ntile = wave & 3;
    const int mtile = (wave >> 2) & 1023;
    const int tp    = wave >> 12;                     // 0..26
    const int hsel  = lane >> 4;                      // half-wave select
    const int lcol  = lane & 15;

    const __bf16* __restrict__ Bw = CW + (size_t)tp * DD * DD;
    const __bf16* __restrict__ Arow =
        evbf + (size_t)(mtile * 16 + lcol) * DD + hsel * 8;
    const int ncol0 = ntile * 64 + lcol;
    const int kboff = hsel * 8;

    v8f acc[4] = {v8f{}, v8f{}, v8f{}, v8f{}};
    for (int ks = 0; ks < 8; ++ks) {
        const int k0 = ks * 32;
        v16bf a = load_a_frag(Arow, k0);
#pragma unroll
        for (int n = 0; n < 4; ++n) {
            v16bf b = load_a_frag(Bw + (size_t)(ncol0 + n * 16) * DD + kboff, k0);
            acc[n] = __builtin_amdgcn_wmma_f32_16x16x32_bf16(
                false, a, false, b, (short)0, acc[n], false, false);
        }
    }
    const size_t outbase =
        ((size_t)tp * NE + (size_t)(mtile * 16 + hsel * 8)) * DD;
#pragma unroll
    for (int n = 0; n < 4; ++n) {
        const int col = ncol0 + n * 16;
#pragma unroll
        for (int r = 0; r < 8; ++r)
            qvg[outbase + (size_t)r * DD + col] = (__bf16)acc[n][r];
    }
}

// ---------------------------------------------------------------------------
// Stage 2: ragged gated scan in (patch,time) row order.
//   h_m = g_m*h_{m-1} + (1-g_m)*v_m, reset where seg_start; u = q*h.
// 256 threads = 256 channels; each block owns 64 rows, back-walks to the
// nearest segment start (segments average ~9.6 rows) to seed h.
// Writes U[(p*NE+e)*D + c] = u  (each (p,e) pair occurs exactly once).
// ---------------------------------------------------------------------------
__global__ __launch_bounds__(256) void gated_scan_kernel(
    const __bf16* __restrict__ qvg,
    const int* __restrict__ orig, const int* __restrict__ pos,
    const unsigned char* __restrict__ segs,   // jnp bool_ -> 1 byte
    __bf16* __restrict__ U) {
    const int c = threadIdx.x;
    const int chunk0 = blockIdx.x * 64;
    const size_t ST = (size_t)PP * NE * DD;   // per-family stride in qvg

    int s = chunk0;
    while (s > 0 && segs[s] == 0) --s;        // uniform back-walk (hw broadcast)

    int mend = chunk0 + 64;
    if (mend > MR) mend = MR;
    float h = 0.f;
    for (int m = s; m < mend; ++m) {
        const size_t pe = ((size_t)pos[m] * NE + (size_t)orig[m]) * DD + c;
        const float gl = (float)qvg[2 * ST + pe];
        const float g  = 1.f / (1.f + __expf(-gl));
        const float v  = (float)qvg[ST + pe];
        h = segs[m] ? (1.f - g) * v : g * h + (1.f - g) * v;
        if (m >= chunk0) {
            const float q = (float)qvg[pe];
            U[pe] = (__bf16)(q * h);
        }
    }
}

// ---------------------------------------------------------------------------
// Stage 3: pre[e,:] = events[e,:] + sum_p U[p] @ CW[27+p]^T
// One wave per 16x64 tile, accumulating 9 positions x 8 ksteps of WMMA.
// ---------------------------------------------------------------------------
__global__ __launch_bounds__(256) void gemm_out_kernel(
    const __bf16* __restrict__ U, const __bf16* __restrict__ CW,
    const float* __restrict__ events, float* __restrict__ pre) {
    const int wave = (blockIdx.x << 3) | (threadIdx.x >> 5);
    const int lane = threadIdx.x & 31;
    const int ntile = wave & 3;
    const int mtile = wave >> 2;                      // 0..1023
    const int hsel  = lane >> 4;
    const int lcol  = lane & 15;
    const int ncol0 = ntile * 64 + lcol;
    const int kboff = hsel * 8;
    const int arow  = mtile * 16 + lcol;

    v8f acc[4] = {v8f{}, v8f{}, v8f{}, v8f{}};
    for (int p = 0; p < PP; ++p) {
        const __bf16* __restrict__ Ap =
            U + ((size_t)p * NE + arow) * DD + hsel * 8;
        const __bf16* __restrict__ Bw = CW + (size_t)(27 + p) * DD * DD;
        for (int ks = 0; ks < 8; ++ks) {
            const int k0 = ks * 32;
            v16bf a = load_a_frag(Ap, k0);
#pragma unroll
            for (int n = 0; n < 4; ++n) {
                v16bf b = load_a_frag(Bw + (size_t)(ncol0 + n * 16) * DD + kboff, k0);
                acc[n] = __builtin_amdgcn_wmma_f32_16x16x32_bf16(
                    false, a, false, b, (short)0, acc[n], false, false);
            }
        }
    }
    const size_t rbase = (size_t)(mtile * 16 + hsel * 8) * DD;
#pragma unroll
    for (int n = 0; n < 4; ++n) {
        const int col = ncol0 + n * 16;
#pragma unroll
        for (int r = 0; r < 8; ++r) {
            const size_t idx = rbase + (size_t)r * DD + col;
            pre[idx] = acc[n][r] + events[idx];       // residual
        }
    }
}

// ---------------------------------------------------------------------------
// LayerNorm: one wave per row (256 channels, 8 per lane), shfl_xor reduce.
// ---------------------------------------------------------------------------
__global__ __launch_bounds__(256) void layernorm_kernel(
    const float* __restrict__ pre, const float* __restrict__ gamma,
    const float* __restrict__ beta, float* __restrict__ out) {
    const int lane = threadIdx.x & 31;
    const int row  = blockIdx.x * 8 + (threadIdx.x >> 5);
    const float* __restrict__ r = pre + (size_t)row * DD + lane * 8;
    float x[8];
#pragma unroll
    for (int j = 0; j < 8; ++j) x[j] = r[j];
    float s = 0.f, ss = 0.f;
#pragma unroll
    for (int j = 0; j < 8; ++j) { s += x[j]; ss += x[j] * x[j]; }
#pragma unroll
    for (int off = 16; off >= 1; off >>= 1) {
        s  += __shfl_xor(s, off, 32);
        ss += __shfl_xor(ss, off, 32);
    }
    const float mu  = s * (1.f / DD);
    const float var = ss * (1.f / DD) - mu * mu;
    const float rs  = __frsqrt_rn(var + 1e-5f);
    float* __restrict__ o = out + (size_t)row * DD + lane * 8;
#pragma unroll
    for (int j = 0; j < 8; ++j) {
        const int c = lane * 8 + j;
        o[j] = (x[j] - mu) * rs * gamma[c] + beta[c];
    }
}

// ---------------------------------------------------------------------------
// Workspace layout (bytes)
// ---------------------------------------------------------------------------
#define WS_EVBF 0u
#define WS_CW   (WS_EVBF + (size_t)NE * DD * 2)                 //  8,388,608
#define WS_QVG  (WS_CW   + (size_t)36 * DD * DD * 2)            // +4,718,592
#define WS_U    (WS_QVG  + (size_t)3 * PP * NE * DD * 2)        // +226,492,416
#define WS_PRE  (WS_U    + (size_t)PP * NE * DD * 2)            // +75,497,472
// total = WS_PRE + NE*DD*4  ~= 331.9 MB

extern "C" void kernel_launch(void* const* d_in, const int* in_sizes, int n_in,
                              void* d_out, int out_size, void* d_ws, size_t ws_size,
                              hipStream_t stream) {
    const float* events  = (const float*)d_in[0];
    const float* scatter = (const float*)d_in[1];
    const float* gather  = (const float*)d_in[2];
    const float* q_w     = (const float*)d_in[3];
    const float* v_w     = (const float*)d_in[4];
    const float* g_w     = (const float*)d_in[5];
    const float* o_w     = (const float*)d_in[6];
    const float* ln_g    = (const float*)d_in[7];
    const float* ln_b    = (const float*)d_in[8];
    const int*   orig    = (const int*)d_in[9];
    const int*   pos     = (const int*)d_in[10];
    const unsigned char* segs = (const unsigned char*)d_in[11];
    float* out = (float*)d_out;

    char* ws = (char*)d_ws;
    __bf16* evbf = (__bf16*)(ws + WS_EVBF);
    __bf16* CW   = (__bf16*)(ws + WS_CW);
    __bf16* qvg  = (__bf16*)(ws + WS_QVG);
    __bf16* U    = (__bf16*)(ws + WS_U);
    float*  pre  = (float*)(ws + WS_PRE);

    // 0) events -> bf16
    cvt_bf16_kernel<<<(NE * DD) / 256, 256, 0, stream>>>(events, evbf, NE * DD);

    // 1) fuse the 36 combined 256x256 weight matrices (bf16, B-layout)
    fuse_weights_kernel<<<dim3(DD, 36), 256, 0, stream>>>(
        q_w, v_w, g_w, o_w, scatter, gather, CW);

    // 2) 27 dense GEMMs: qvg = events @ CW[t,p]^T  (110592 wave-tiles)
    gemm_qvg_kernel<<<(27 * 1024 * 4) / 8, 256, 0, stream>>>(evbf, CW, qvg);

    // 3) ragged gated scan -> U = q * h
    gated_scan_kernel<<<MR / 64, 256, 0, stream>>>(qvg, orig, pos, segs, U);

    // 4) out-projection GEMM accumulated over 9 positions + residual
    gemm_out_kernel<<<(1024 * 4) / 8, 256, 0, stream>>>(U, CW, events, pre);

    // 5) LayerNorm
    layernorm_kernel<<<NE / 8, 256, 0, stream>>>(pre, ln_g, ln_b, out);
}
